// SimpleSelfAttention_32925219291325
// MI455X (gfx1250) — matrix operand order repaired and checked by reading, not compiled, not run on hardware
//
#include <hip/hip_runtime.h>

typedef __attribute__((ext_vector_type(16))) _Float16 v16h;
typedef __attribute__((ext_vector_type(8)))  _Float16 v8h;
typedef __attribute__((ext_vector_type(8)))  float    v8f;

#define WMMA_F32_F16(a, b, c) \
  __builtin_amdgcn_wmma_f32_16x16x32_f16(false, (a), false, (b), (short)0, (c), false, false)

static constexpr int Bn = 4, Sn = 4096, Dn = 256;

// -------------------------------------------------------------------------
// Kernel 0: one-time W -> f16 transpose.  Wt[mat][n][k] = (f16) W[k][n].
// 3*256*256 elements; reads coalesced along n, one-time scattered writes.
// -------------------------------------------------------------------------
__global__ __launch_bounds__(256)
void convert_w_kernel(const float* __restrict__ Wq,
                      const float* __restrict__ Wk,
                      const float* __restrict__ Wv,
                      _Float16* __restrict__ Wt)
{
  const int idx = blockIdx.x * 256 + threadIdx.x;   // [0, 3*65536)
  const int mat = idx >> 16;
  const int k   = (idx >> 8) & 255;                 // row of W
  const int n   = idx & 255;                        // col of W
  const float* W = (mat == 0) ? Wq : (mat == 1) ? Wk : Wv;
  Wt[((size_t)mat << 16) + ((size_t)n << 8) + k] =
      (_Float16)W[((size_t)k << 8) + n];
}

// -------------------------------------------------------------------------
// Kernel 1: QKV projection.  grid = (B*S/16, 3 {Q,K,V}), block = 32 (1 wave).
// Each wave computes a 16-row x 256-col output strip with f16 WMMA, f32 acc.
// B-fragments come from Wt (f16, transposed) as contiguous 32-byte loads.
// Q,K stored row-major f16 [B*S][D]; V stored transposed f16 [B][D][S].
// -------------------------------------------------------------------------
__global__ __launch_bounds__(32)
void qkv_proj_kernel(const float* __restrict__ x,
                     const _Float16* __restrict__ Wt,
                     const float* __restrict__ bq,
                     const float* __restrict__ bk,
                     const float* __restrict__ bv,
                     _Float16* __restrict__ Qo, _Float16* __restrict__ Ko,
                     _Float16* __restrict__ Vt)
{
  const int lane = threadIdx.x & 31;
  const int nm   = lane & 15;   // A-row / B-col / C-col for this lane
  const int half = lane >> 4;
  const int r0   = blockIdx.x * 16;     // global row tile over [B*S]
  const int mat  = blockIdx.y;          // 0=Q 1=K 2=V

  const _Float16* Wm  = Wt + ((size_t)mat << 16);
  const float* bias   = (mat == 0) ? bq : (mat == 1) ? bk : bv;

  // Preload A fragments of x rows (f32 -> f16), reused across all 16 n-tiles.
  v16h afr[8];
  const float* xr = x + (size_t)(r0 + nm) * Dn;
  #pragma unroll
  for (int kf = 0; kf < 8; ++kf) {
    const float* p = xr + kf * 32 + half * 16;
    v16h a;
    #pragma unroll
    for (int j = 0; j < 8; ++j) a[j] = (_Float16)p[j];
    #pragma unroll
    for (int j = 0; j < 8; ++j) a[8 + j] = (_Float16)p[8 + j];
    afr[kf] = a;
  }

  for (int n0 = 0; n0 < Dn; n0 += 16) {
    const int ncol = n0 + nm;
    const _Float16* wr = Wm + ((size_t)ncol << 8);  // Wt row for this column
    v8f c = {};
    #pragma unroll
    for (int kf = 0; kf < 8; ++kf) {
      // B fragment: b[j] = W[kf*32 + half*16 + j][ncol] = wr[kf*32 + half*16 + j]
      v16h bfr = *(const v16h*)(wr + kf * 32 + half * 16);
      c = WMMA_F32_F16(afr[kf], bfr, c);
    }
    const float bb = bias[ncol];
    #pragma unroll
    for (int i = 0; i < 8; ++i) c[i] += bb;

    if (mat < 2) {
      _Float16* dst = (mat == 0) ? Qo : Ko;
      #pragma unroll
      for (int i = 0; i < 8; ++i) {
        const int row = r0 + i + half * 8;
        dst[(size_t)row * Dn + ncol] = (_Float16)c[i];
      }
    } else {  // V transposed: [B][D][S]
      #pragma unroll
      for (int i = 0; i < 8; ++i) {
        const int row = r0 + i + half * 8;
        const int b = row >> 12;            // S = 4096
        const int s = row & (Sn - 1);
        Vt[((size_t)b * Dn + ncol) * Sn + s] = (_Float16)c[i];
      }
    }
  }
}

// -------------------------------------------------------------------------
// Kernel 2: flash attention.  grid = B*S/16/4, block = 128 (4 waves).
// Each wave owns a 16-query tile, streams keys 32 at a time:
//   S = Q K^T (16 wmma), online softmax, O += P V (16 wmma).
// P transposed C-layout -> A-layout through wave-private LDS.
// -------------------------------------------------------------------------
__global__ __launch_bounds__(128)
void flash_attn_kernel(const _Float16* __restrict__ Q,
                       const _Float16* __restrict__ K,
                       const _Float16* __restrict__ Vt,
                       float* __restrict__ out)
{
  __shared__ __align__(32) _Float16 ldsP[4][16 * 32];
  const int lane = threadIdx.x & 31;
  const int wv   = threadIdx.x >> 5;
  const int nm   = lane & 15;
  const int half = lane >> 4;
  const int r0   = (blockIdx.x * 4 + wv) * 16;   // global query row tile
  const int b    = r0 >> 12;                     // batch index
  const float scale = 0.0625f;                   // 1/sqrt(256)

  // Q A-fragments (f16, register-resident for the whole key loop)
  v16h qa[8];
  const _Float16* qr = Q + (size_t)(r0 + nm) * Dn;
  #pragma unroll
  for (int kf = 0; kf < 8; ++kf) {
    v8h lo = *(const v8h*)(qr + kf * 32 + half * 16);
    v8h hi = *(const v8h*)(qr + kf * 32 + half * 16 + 8);
    v16h a;
    #pragma unroll
    for (int j = 0; j < 8; ++j) { a[j] = lo[j]; a[8 + j] = hi[j]; }
    qa[kf] = a;
  }

  v8f o[16];
  const v8f vzero = {};
  #pragma unroll
  for (int nt = 0; nt < 16; ++nt) o[nt] = vzero;
  float mrow[8], lrow[8];
  #pragma unroll
  for (int i = 0; i < 8; ++i) { mrow[i] = -1e30f; lrow[i] = 0.0f; }

  for (int kb = 0; kb < Sn; kb += 32) {
    // --- score tiles: s0 = keys kb..kb+15, s1 = keys kb+16..kb+31 ---
    const _Float16* kp0 = K + (size_t)(b * Sn + kb + nm) * Dn;
    const _Float16* kp1 = kp0 + (size_t)16 * Dn;
    v8f s0 = {}, s1 = {};
    #pragma unroll
    for (int kf = 0; kf < 8; ++kf) {
      v16h b0 = *(const v16h*)(kp0 + kf * 32 + half * 16);
      v16h b1 = *(const v16h*)(kp1 + kf * 32 + half * 16);
      s0 = WMMA_F32_F16(qa[kf], b0, s0);
      s1 = WMMA_F32_F16(qa[kf], b1, s1);
    }

    // --- online softmax (row stats via intra-half butterfly reductions) ---
    #pragma unroll
    for (int i = 0; i < 8; ++i) {
      float a0 = s0[i] * scale, a1 = s1[i] * scale;
      float tm = fmaxf(a0, a1);
      tm = fmaxf(tm, __shfl_xor(tm, 1));
      tm = fmaxf(tm, __shfl_xor(tm, 2));
      tm = fmaxf(tm, __shfl_xor(tm, 4));
      tm = fmaxf(tm, __shfl_xor(tm, 8));
      const float nmax  = fmaxf(mrow[i], tm);
      const float alpha = __expf(mrow[i] - nmax);
      const float p0 = __expf(a0 - nmax);
      const float p1 = __expf(a1 - nmax);
      float rs = p0 + p1;
      rs += __shfl_xor(rs, 1);
      rs += __shfl_xor(rs, 2);
      rs += __shfl_xor(rs, 4);
      rs += __shfl_xor(rs, 8);
      lrow[i] = lrow[i] * alpha + rs;
      mrow[i] = nmax;
      // stage P (C-layout -> LDS, row = i + half*8)
      ldsP[wv][(i + half * 8) * 32 + nm]      = (_Float16)p0;
      ldsP[wv][(i + half * 8) * 32 + 16 + nm] = (_Float16)p1;
      #pragma unroll
      for (int nt = 0; nt < 16; ++nt) o[nt][i] *= alpha;
    }

    // --- reload P as A fragment (wave-private LDS; DS ops in-order) ---
    v16h pa;
    {
      const _Float16* pp = &ldsP[wv][nm * 32];
      v8h lo = *(const v8h*)(pp + half * 16);
      v8h hi = *(const v8h*)(pp + half * 16 + 8);
      #pragma unroll
      for (int j = 0; j < 8; ++j) { pa[j] = lo[j]; pa[8 + j] = hi[j]; }
    }

    // --- O += P @ V  (V^T gives contiguous B fragments) ---
    #pragma unroll
    for (int nt = 0; nt < 16; ++nt) {
      const _Float16* vp =
          Vt + ((size_t)b * Dn + nt * 16 + nm) * Sn + kb + half * 16;
      v16h bvf = *(const v16h*)vp;
      o[nt] = WMMA_F32_F16(pa, bvf, o[nt]);
    }
  }

  // --- epilogue: out = O / l ---
  #pragma unroll
  for (int i = 0; i < 8; ++i) {
    const float inv = 1.0f / lrow[i];
    const int row = r0 + i + half * 8;
    #pragma unroll
    for (int nt = 0; nt < 16; ++nt)
      out[(size_t)row * Dn + nt * 16 + nm] = o[nt][i] * inv;
  }
}

// -------------------------------------------------------------------------
extern "C" void kernel_launch(void* const* d_in, const int* in_sizes, int n_in,
                              void* d_out, int out_size, void* d_ws, size_t ws_size,
                              hipStream_t stream) {
  const float* x  = (const float*)d_in[0];
  const float* Wq = (const float*)d_in[1];
  const float* bq = (const float*)d_in[2];
  const float* Wk = (const float*)d_in[3];
  const float* bk = (const float*)d_in[4];
  const float* Wv = (const float*)d_in[5];
  const float* bv = (const float*)d_in[6];
  float* out = (float*)d_out;

  // workspace: Q (8MB) | K (8MB) | V^T (8MB) | W^T (384KB), all f16
  _Float16* Qh = (_Float16*)d_ws;
  _Float16* Kh = Qh + (size_t)Bn * Sn * Dn;
  _Float16* Vt = Kh + (size_t)Bn * Sn * Dn;
  _Float16* Wt = Vt + (size_t)Bn * Sn * Dn;

  convert_w_kernel<<<3 * Dn * Dn / 256, 256, 0, stream>>>(Wq, Wk, Wv, Wt);

  dim3 gridP(Bn * Sn / 16, 3);
  qkv_proj_kernel<<<gridP, 32, 0, stream>>>(x, Wt, bq, bk, bv, Qh, Kh, Vt);

  dim3 gridA(Bn * Sn / 16 / 4);
  flash_attn_kernel<<<gridA, 128, 0, stream>>>(Qh, Kh, Vt, out);
}
